// AdvancedHopfieldLayer_2731599200529
// MI455X (gfx1250) — matrix operand order repaired and checked by reading, not compiled, not run on hardware
//
#include <hip/hip_runtime.h>
#include <hip/hip_bf16.h>
#include <math.h>

// CDNA5 / gfx1250 (MI455X), wave32. WMMA bf16 16x16x32 for the connectivity
// matvec iterations. x materialized once in bf16 (row-major + transposed,
// 64 MB total; L2 is 192 MB so iterations run out of L2). The transposed
// copy makes every WMMA B fragment 16 *contiguous* bf16 -> 2x b128 loads.

typedef __attribute__((ext_vector_type(16))) __bf16 v16bf;
typedef __attribute__((ext_vector_type(8)))  float  v8f;

#define TPB 256

// ---------------- block reductions (blockDim.x == TPB) ----------------
__device__ inline float blk_sum(float v, float* s) {
    int t = threadIdx.x;
    s[t] = v; __syncthreads();
    for (int o = TPB / 2; o > 0; o >>= 1) {
        if (t < o) s[t] += s[t + o];
        __syncthreads();
    }
    float r = s[0]; __syncthreads();
    return r;
}
__device__ inline float blk_max(float v, float* s) {
    int t = threadIdx.x;
    s[t] = v; __syncthreads();
    for (int o = TPB / 2; o > 0; o >>= 1) {
        if (t < o) s[t] = fmaxf(s[t], s[t + o]);
        __syncthreads();
    }
    float r = s[0]; __syncthreads();
    return r;
}

// ---------------- pass 1: fused row softmax/sigmoid/mask + stats -------
// One block per row. acc layout: [0]=path_cost [1]=binary [2]=n_edges
//                                [3]=flow_pen  [4]=sum_x
__global__ void row_pass(const float* __restrict__ dist,
                         const float* __restrict__ logits,
                         const float* __restrict__ att,
                         __bf16* __restrict__ xbf,
                         float* __restrict__ out_flow,
                         float* __restrict__ acc, int n) {
    __shared__ float s_row[4096];
    __shared__ float s_red[TPB];
    int row = blockIdx.x, tid = threadIdx.x;
    size_t base = (size_t)row * n;

    float m = -3.4e38f;
    for (int j = tid; j < n; j += TPB) {
        float a = att[base + j];
        s_row[j] = a;
        m = fmaxf(m, a);
    }
    float rmax = blk_max(m, s_red);

    float se = 0.f;
    for (int j = tid; j < n; j += TPB) se += __expf(s_row[j] - rmax);
    float inv = 1.f / blk_sum(se, s_red);

    float rs = 0.f, pc = 0.f, bp = 0.f, ne = 0.f;
    for (int j = tid; j < n; j += TPB) {
        float a  = __expf(s_row[j] - rmax) * inv;            // softmax
        float dg = dist[base + j];
        float lg = logits[base + j];
        float sg = 1.f / (1.f + __expf(-2.f * lg));          // sigmoid(l/0.5)
        bool  ok = dg < 1e6f;
        float xv = ok ? sg * a : 0.f;
        xbf[base + j] = (__bf16)xv;
        rs += xv;
        bp += xv * (1.f - xv);
        if (ok) { ne += 1.f; pc += dg * xv; }
    }
    rs = blk_sum(rs, s_red);
    pc = blk_sum(pc, s_red);
    bp = blk_sum(bp, s_red);
    ne = blk_sum(ne, s_red);
    if (tid == 0) {
        out_flow[row] = rs;
        atomicAdd(&acc[0], pc);
        atomicAdd(&acc[1], bp);
        atomicAdd(&acc[2], ne);
    }
}

// ---------------- transpose: xT[c][r] = x[r][c] (64x64 LDS tiles) ------
__global__ void transpose_bf(const __bf16* __restrict__ x,
                             __bf16* __restrict__ xT, int n) {
    __shared__ __bf16 tile[64][65];            // +1 pad vs bank conflicts
    int t = threadIdx.x;
    int tiles = n / 64;
    int bx = blockIdx.x % tiles;               // column tile
    int by = blockIdx.x / tiles;               // row tile
    int c0 = bx * 64, r0 = by * 64;
    int lc = t & 63, lr4 = t >> 6;             // 4 rows per step
#pragma unroll
    for (int i = 0; i < 16; ++i) {
        int lr = lr4 + i * 4;
        tile[lr][lc] = x[(size_t)(r0 + lr) * n + c0 + lc];
    }
    __syncthreads();
#pragma unroll
    for (int i = 0; i < 16; ++i) {
        int lr = lr4 + i * 4;                  // lr = original column index
        xT[(size_t)(c0 + lr) * n + r0 + lc] = tile[lc][lr];
    }
}

// ---------------- pass 2: column sums (in_flow), row-major x -----------
__global__ void col_sum(const __bf16* __restrict__ x,
                        float* __restrict__ in_flow, int n) {
    int colblocks = n / TPB;
    int cb = blockIdx.x % colblocks;
    int rsl = blockIdx.x / colblocks;          // 16 row slices
    int c = cb * TPB + threadIdx.x;
    int rows = n / 16;
    int r0 = rsl * rows;
    float a = 0.f;
    const __bf16* p = x + (size_t)r0 * n + c;
    for (int r = 0; r < rows; ++r) { a += (float)(*p); p += n; }
    atomicAdd(&in_flow[c], a);
}

// ---------------- prep: flow penalty, sum_x, r0 ------------------------
__global__ void prep(const float* __restrict__ of, const float* __restrict__ inf_,
                     const __bf16* __restrict__ x,
                     const int* __restrict__ psrc, const int* __restrict__ pdst,
                     float* __restrict__ r0, __bf16* __restrict__ r0b,
                     float* __restrict__ acc, int n) {
    __shared__ float s_red[TPB];
    int i = blockIdx.x * blockDim.x + threadIdx.x;
    int src = *psrc, dst = *pdst;
    float d = of[i] - inf_[i] + (i == src ? -1.f : 0.f) + (i == dst ? 1.f : 0.f);
    __bf16 xv = x[(size_t)src * n + i];
    r0[i] = (float)xv;
    r0b[i] = xv;
    float fp = blk_sum(d * d, s_red);
    float sx = blk_sum(of[i], s_red);
    if (threadIdx.x == 0) { atomicAdd(&acc[3], fp); atomicAdd(&acc[4], sx); }
}

// ---------------- connectivity matvec: t += r @ x via WMMA bf16 --------
// A (16x32 bf16) = r-chunk replicated into all 16 rows -> all D rows equal.
// B (32x16 bf16) = x[k0:k0+32, j0:j0+16], read from xT so each lane's 16
// K-elements are contiguous: two b128 loads, k = 16*(lane/16) + elem, same order as A.
// Contiguous K order in A and B (not the ISA table order); M/N = lane&15.
union Frag { uint4 u[2]; v16bf v; };

template <int NSTAT>
__global__ void matvec_wmma(const __bf16* __restrict__ xT,
                            const __bf16* __restrict__ rbf,
                            float* __restrict__ t, int ndyn) {
    const int n = (NSTAT > 0) ? NSTAT : ndyn;
    const int KSPLIT = 16;
    int lane = threadIdx.x & 31;
    int wave = threadIdx.x >> 5;               // 0..7, each owns 16 columns
    int colblocks = n / 128;                   // 32
    int cb = blockIdx.x % colblocks;
    int ks = blockIdx.x / colblocks;           // 0..15 k-slices
    int kper = n / KSPLIT;                     // 256
    int kbeg = ks * kper;
    int j0 = cb * 128 + wave * 16;

    int nsel  = lane & 15;                     // B: N = lane&15
    int kbase = (lane < 16) ? 0 : 16;          // B: K window per half-wave
    int abase = (lane < 16) ? 0 : 16;          // A: K window per half-wave

    // xT layout: xT[j*n + k]. Both bases are 32B-aligned (kbeg%256==0).
    const __bf16* bp = xT + (size_t)(j0 + nsel) * n + kbeg + kbase;
    const __bf16* rp = rbf + kbeg + abase;

    v8f c = {};
    for (int kt = 0; kt < kper; kt += 32) {
        Frag A, B;
        B.u[0] = *(const uint4*)(bp);          // B elems 0..7  (K kbase+0..7)
        B.u[1] = *(const uint4*)(bp + 8);      // B elems 8..15 (K kbase+8..15)
        A.u[0] = *(const uint4*)(rp);          // A elems 0..7  (K abase+0..7)
        A.u[1] = *(const uint4*)(rp + 8);      // A elems 8..15 (K abase+8..15)
        if (kt + 32 < kper) __builtin_prefetch(bp + 1024, 0, 0);
        // (neg_a, A, neg_b, B, c_mod, C, reuse_a, reuse_b)
        c = __builtin_amdgcn_wmma_f32_16x16x32_bf16(false, A.v, false, B.v,
                                                    (short)0, c, false, false);
        bp += 32;
        rp += 32;
    }
    // All D rows identical; row M=0 lives in VGPR0 of lanes 0..15.
    if (lane < 16) atomicAdd(&t[j0 + lane], c[0]);
}

// ---------------- r update: r = min(r + t, 1); reset t -----------------
__global__ void r_update(const float* __restrict__ rin, float* __restrict__ t,
                         float* __restrict__ rout, __bf16* __restrict__ routb,
                         int n) {
    int i = blockIdx.x * blockDim.x + threadIdx.x;
    float v = fminf(rin[i] + t[i], 1.f);
    rout[i] = v;
    routb[i] = (__bf16)v;
    t[i] = 0.f;                                // ready for next iteration
}

// ---------------- finalize ---------------------------------------------
__global__ void finalize(const float* __restrict__ acc,
                         const float* __restrict__ rfin,
                         const int* __restrict__ pdst, int n,
                         float* __restrict__ out) {
    if (threadIdx.x == 0 && blockIdx.x == 0) {
        float pc = acc[0], bp = acc[1], ne = acc[2], fp = acc[3], sx = acc[4];
        float nn = (float)n * (float)n;
        float density = ne / nn;
        float mu23 = 10.f * (1.f + density);
        float conn = 1.f - rfin[*pdst];
        conn *= conn;
        out[0] = pc / (ne + 1e-6f)
               + mu23 * fp / (float)n
               + mu23 * bp / nn
               + 20.f * conn
               + 5.f * sx / nn;
    }
}

extern "C" void kernel_launch(void* const* d_in, const int* in_sizes, int n_in,
                              void* d_out, int out_size, void* d_ws, size_t ws_size,
                              hipStream_t stream) {
    (void)n_in; (void)out_size; (void)ws_size;
    const float* dist   = (const float*)d_in[0];
    const float* logits = (const float*)d_in[1];
    const float* att    = (const float*)d_in[2];
    const int*   psrc   = (const int*)d_in[3];
    const int*   pdst   = (const int*)d_in[4];

    int n = (int)(sqrt((double)in_sizes[0]) + 0.5);   // 4096

    char* p = (char*)d_ws;
    __bf16* xbf     = (__bf16*)p; p += (size_t)n * n * sizeof(__bf16);
    __bf16* xT      = (__bf16*)p; p += (size_t)n * n * sizeof(__bf16);
    float*  out_fl  = (float*)p;  p += (size_t)n * 4;
    float*  in_fl   = (float*)p;  p += (size_t)n * 4;
    float*  tvec    = (float*)p;  p += (size_t)n * 4;
    float*  ra      = (float*)p;  p += (size_t)n * 4;
    float*  rb      = (float*)p;  p += (size_t)n * 4;
    __bf16* rabf    = (__bf16*)p; p += (size_t)n * 2;
    __bf16* rbbf    = (__bf16*)p; p += (size_t)n * 2;
    float*  acc     = (float*)p;  p += 64;

    hipMemsetAsync(acc, 0, 64, stream);
    hipMemsetAsync(in_fl, 0, (size_t)n * 4, stream);
    hipMemsetAsync(tvec, 0, (size_t)n * 4, stream);

    row_pass<<<n, TPB, 0, stream>>>(dist, logits, att, xbf, out_fl, acc, n);
    transpose_bf<<<(n / 64) * (n / 64), TPB, 0, stream>>>(xbf, xT, n);
    col_sum<<<(n / TPB) * 16, TPB, 0, stream>>>(xbf, in_fl, n);
    prep<<<n / TPB, TPB, 0, stream>>>(out_fl, in_fl, xbf, psrc, pdst, ra, rabf, acc, n);

    float*  rc = ra;  float*  rn = rb;
    __bf16* rcb = rabf; __bf16* rnb = rbbf;
    for (int it = 0; it < 10; ++it) {
        if (n == 4096)
            matvec_wmma<4096><<<(n / 128) * 16, TPB, 0, stream>>>(xT, rcb, tvec, n);
        else
            matvec_wmma<0><<<(n / 128) * 16, TPB, 0, stream>>>(xT, rcb, tvec, n);
        r_update<<<n / TPB, TPB, 0, stream>>>(rc, tvec, rn, rnb, n);
        float* tf = rc; rc = rn; rn = tf;
        __bf16* tb = rcb; rcb = rnb; rnb = tb;
    }
    finalize<<<1, 32, 0, stream>>>(acc, rc, pdst, n, (float*)d_out);
}
